// NeuronTrinityDecoderLayer_80376017977949
// MI455X (gfx1250) — compile-verified, matrix operand order checked by reading, not run on hardware
//
#include <hip/hip_runtime.h>

// ---------------------------------------------------------------------------
// Types for CDNA5 WMMA (wave32). v_wmma_f32_16x16x32_bf16: D(16x16 f32) =
// A(16x32 bf16) * B(32x16 bf16) + C.
// ---------------------------------------------------------------------------
typedef __attribute__((ext_vector_type(16))) __bf16 v16bf;
typedef __attribute__((ext_vector_type(8)))  float  v8f;
typedef int v4i_vs __attribute__((vector_size(16)));   // matches builtin param

constexpr int kT    = 512;               // tokens (B*S)
constexpr int kH    = 2048;              // hidden
constexpr int kNH   = 32;                // attention heads
constexpr int kKVH  = 4;                 // kv heads
constexpr int kHD   = 128;               // head dim
constexpr int kQD   = kNH * kHD;         // 4096
constexpr int kKD   = kKVH * kHD;        // 512
constexpr int kE    = 32;                // experts
constexpr int kTOPK = 8;
constexpr int kI    = 1024;              // moe intermediate
constexpr int kA    = kT * kTOPK;        // total assignments = 4096
constexpr float kEPS = 1e-5f;

// ---------------------------------------------------------------------------
// Async global->LDS (CDNA5 GLOBAL_LOAD_ASYNC_TO_LDS_B128, ASYNCcnt), guarded
// so a missing builtin falls back to a synchronous 16-byte copy.
// Builtin signature (from compiler diagnostic): param0 = v4i* in __device__
// (AS1 global) space, non-const; param1 = v4i* LDS; then imm offset, imm cpol.
// ---------------------------------------------------------------------------
#if defined(__HIP_DEVICE_COMPILE__) && __has_builtin(__builtin_amdgcn_global_load_async_to_lds_b128)
#define USE_ASYNC_LDS 1
#endif

__device__ __forceinline__ void cp16_g2l(__bf16* dst_lds, const __bf16* src_g) {
#ifdef USE_ASYNC_LDS
  __builtin_amdgcn_global_load_async_to_lds_b128(
      (__attribute__((address_space(1))) v4i_vs*)(v4i_vs*)src_g,
      (__attribute__((address_space(3))) v4i_vs*)(v4i_vs*)dst_lds,
      /*offset=*/0, /*cpol=*/0);
#else
  *(uint4*)dst_lds = *(const uint4*)src_g;
#endif
}

__device__ __forceinline__ void wait_async_lds() {
#ifdef USE_ASYNC_LDS
#if __has_builtin(__builtin_amdgcn_s_wait_asynccnt)
  __builtin_amdgcn_s_wait_asynccnt(0);
#else
  asm volatile("s_wait_asynccnt 0" ::: "memory");
#endif
#endif
}

// ---------------------------------------------------------------------------
// WMMA helpers
// ---------------------------------------------------------------------------
__device__ __forceinline__ v8f zero8() {
  v8f c;
#pragma unroll
  for (int i = 0; i < 8; ++i) c[i] = 0.0f;
  return c;
}

__device__ __forceinline__ v8f wmma_bf16(v16bf a, v16bf b, v8f c) {
  return __builtin_amdgcn_wmma_f32_16x16x32_bf16(
      /*neg_a=*/false, a, /*neg_b=*/false, b,
      /*c_mod=*/(short)0, c, /*reuse_a=*/false, /*reuse_b=*/false);
}

// Load an A-style fragment (also used for B since we store B N-major):
// 16 rows x 32 K, row stride `ld` bf16 elements. ISA layout: lane l holds row
// l%16; lanes 0-15 take K {0..7,16..23}, lanes 16-31 take K {8..15,24..31}.
__device__ __forceinline__ v16bf ld_frag(const __bf16* base, int ld) {
  const int lane = threadIdx.x & 31;
  const int r    = lane & 15;
  const int hf   = lane >> 4;
  const __bf16* p = base + r * ld + hf * 8;
  v16bf a;
#pragma unroll
  for (int j = 0; j < 8; ++j) {
    a[j]     = p[j];
    a[8 + j] = p[16 + j];
  }
  return a;
}

// ---------------------------------------------------------------------------
// Elementwise / small kernels
// ---------------------------------------------------------------------------
__global__ __launch_bounds__(256) void zero_kernel(unsigned int* p, int n) {
  int i = blockIdx.x * 256 + threadIdx.x;
  if (i < n) p[i] = 0u;
}

__global__ __launch_bounds__(256) void add_kernel(const float* __restrict__ a,
                                                  const float* __restrict__ b,
                                                  float* __restrict__ o, int n) {
  int i = blockIdx.x * 256 + threadIdx.x;
  if (i < n) o[i] = a[i] + b[i];
}

__global__ __launch_bounds__(256) void rmsnorm_kernel(const float* __restrict__ x,
                                                      const float* __restrict__ w,
                                                      __bf16* __restrict__ out) {
  const int t = blockIdx.x;
  __shared__ float red[8];
  float acc = 0.0f;
  for (int i = threadIdx.x; i < kH; i += 256) {
    float v = x[(size_t)t * kH + i];
    acc += v * v;
  }
#pragma unroll
  for (int off = 16; off >= 1; off >>= 1) acc += __shfl_xor(acc, off, 32);
  if ((threadIdx.x & 31) == 0) red[threadIdx.x >> 5] = acc;
  __syncthreads();
  if (threadIdx.x == 0) {
    float s = 0.0f;
#pragma unroll
    for (int i = 0; i < 8; ++i) s += red[i];
    red[0] = rsqrtf(s / (float)kH + kEPS);
  }
  __syncthreads();
  const float rstd = red[0];
  for (int i = threadIdx.x; i < kH; i += 256)
    out[(size_t)t * kH + i] = (__bf16)(x[(size_t)t * kH + i] * rstd * w[i]);
}

__global__ __launch_bounds__(256) void silu_mul_kernel(const float* __restrict__ g,
                                                       const float* __restrict__ u,
                                                       __bf16* __restrict__ out, int n) {
  int i = blockIdx.x * 256 + threadIdx.x;
  if (i >= n) return;
  float gv = g[i];
  float s  = gv * (1.0f / (1.0f + __expf(-gv)));
  out[i]   = (__bf16)(s * u[i]);
}

__global__ __launch_bounds__(64) void rope_kernel(const float* __restrict__ qf,
                                                  const float* __restrict__ kf,
                                                  const float* __restrict__ vf,
                                                  __bf16* __restrict__ qbf,
                                                  __bf16* __restrict__ kbf,
                                                  __bf16* __restrict__ vbf) {
  const int s = blockIdx.x, h = blockIdx.y, i = threadIdx.x;  // i in [0,64)
  const float inv = __powf(10000.0f, -(float)i / 64.0f);
  const float ang = (float)s * inv;
  const float c = __cosf(ang), sn = __sinf(ang);
  {
    const float* qp = qf + ((size_t)s * kNH + h) * kHD;
    __bf16* qo = qbf + ((size_t)s * kNH + h) * kHD;
    float x1 = qp[i], x2 = qp[i + 64];
    qo[i]      = (__bf16)(x1 * c - x2 * sn);
    qo[i + 64] = (__bf16)(x1 * sn + x2 * c);
  }
  if (h < kKVH) {
    const float* kp = kf + ((size_t)s * kKVH + h) * kHD;
    __bf16* ko = kbf + ((size_t)s * kKVH + h) * kHD;
    float x1 = kp[i], x2 = kp[i + 64];
    ko[i]      = (__bf16)(x1 * c - x2 * sn);
    ko[i + 64] = (__bf16)(x1 * sn + x2 * c);
    const float* vp = vf + ((size_t)s * kKVH + h) * kHD;
    __bf16* vo = vbf + ((size_t)s * kKVH + h) * kHD;
    vo[i]      = (__bf16)vp[i];
    vo[i + 64] = (__bf16)vp[i + 64];
  }
}

// ---------------------------------------------------------------------------
// Generic tiled WMMA GEMM: C[MxN] = A_bf16[MxK] @ B_f32[KxN] (f32 accumulate).
// Block: 128 threads = 4 waves; block tile 64x64, K-step 32. Each wave owns a
// 16x64 strip: 1 A fragment reused across 4 B fragments = 4 WMMA / 5 loads.
// A tile streamed via async global->LDS; B converted f32->bf16 through VALU.
// ATOMIC=true -> atomicAdd into C (shared-expert down projection).
// ---------------------------------------------------------------------------
template <bool ATOMIC>
__global__ __launch_bounds__(128) void gemm_bf16_kernel(const __bf16* __restrict__ A,
                                                        const float* __restrict__ B,
                                                        float* __restrict__ C,
                                                        int M, int N, int K) {
  __shared__ __bf16 As[64][40];  // 80B row stride (16B aligned for b128)
  __shared__ __bf16 Bt[64][40];  // N-major: Bt[n][k]
  const int tile_n = blockIdx.x * 64, tile_m = blockIdx.y * 64;
  const int tid = threadIdx.x, wave = tid >> 5, lane = tid & 31;
  const int wm = wave * 16;
  v8f c[4];
#pragma unroll
  for (int j = 0; j < 4; ++j) c[j] = zero8();

  for (int k0 = 0; k0 < K; k0 += 32) {
    if (tid == 0 && k0 + 32 < K)
      __builtin_prefetch(&B[(size_t)(k0 + 32) * N + tile_n], 0, 1);  // global_prefetch_b8
    // A tile: 64 rows x 32 bf16 = 64B/row = 4 x b128 chunks per row.
#pragma unroll
    for (int ch = tid; ch < 256; ch += 128) {
      int r = ch >> 2, cp = ch & 3;
      int gm = tile_m + r; if (gm >= M) gm = M - 1;  // pad rows discarded on store
      cp16_g2l(&As[r][cp * 8], &A[(size_t)gm * K + k0 + cp * 8]);
    }
    // B tile: 64 n x 32 k, f32 -> bf16, coalesced over n.
    for (int i = tid; i < 2048; i += 128) {
      int n = i & 63, kk = i >> 6;
      int gn = tile_n + n;
      Bt[n][kk] = (gn < N) ? (__bf16)B[(size_t)(k0 + kk) * N + gn] : (__bf16)0.0f;
    }
    wait_async_lds();
    __syncthreads();
    v16bf a = ld_frag(&As[wm][0], 40);
#pragma unroll
    for (int j = 0; j < 4; ++j) {
      v16bf b = ld_frag(&Bt[j * 16][0], 40);
      c[j] = wmma_bf16(a, b, c[j]);
    }
    __syncthreads();
  }
  const int r0 = lane & 15, hf = lane >> 4;
#pragma unroll
  for (int j = 0; j < 4; ++j)
#pragma unroll
    for (int r = 0; r < 8; ++r) {
      int gm = tile_m + wm + r + 8 * hf;
      int gn = tile_n + j * 16 + r0;
      if (gm < M && gn < N) {
        if (ATOMIC) atomicAdd(&C[(size_t)gm * N + gn], c[j][r]);
        else        C[(size_t)gm * N + gn] = c[j][r];
      }
    }
}

// ---------------------------------------------------------------------------
// Flash-style causal attention. Grid (S/16, NH), block = 1 wave (32 lanes).
// QK^T and P*V both via WMMA bf16; online softmax with wave16 reductions.
// Q/K tiles streamed via async global->LDS.
// ---------------------------------------------------------------------------
__global__ __launch_bounds__(32) void attn_kernel(const __bf16* __restrict__ qbf,
                                                  const __bf16* __restrict__ kbf,
                                                  const __bf16* __restrict__ vbf,
                                                  __bf16* __restrict__ attnbf) {
  __shared__ __bf16 Qs[16][136];   // 272B stride (16B aligned)
  __shared__ __bf16 Ks[32][136];
  __shared__ __bf16 Vst[128][36];  // transposed: Vst[d][key]
  __shared__ __bf16 Ps[16][36];
  const int qt = blockIdx.x, hh = blockIdx.y;
  const int kvh = hh / (kNH / kKVH);
  const int lane = threadIdx.x, r0 = lane & 15, hf = lane >> 4;

  // Q tile: 16 rows x 128 bf16 = 16 x b128 chunks per row.
  for (int ch = lane; ch < 16 * 16; ch += 32) {
    int r = ch >> 4, cp = ch & 15;
    cp16_g2l(&Qs[r][cp * 8],
             &qbf[((size_t)(qt * 16 + r) * kNH + hh) * kHD + cp * 8]);
  }

  v8f o[8];
#pragma unroll
  for (int nt = 0; nt < 8; ++nt) o[nt] = zero8();
  float mrow[8], lsum[8];
#pragma unroll
  for (int r = 0; r < 8; ++r) { mrow[r] = -1e30f; lsum[r] = 0.0f; }

  const float scale = 0.08838834764831845f;  // 1/sqrt(128)
  const int qmax = qt * 16 + 15;
  const int nkb = qmax / 32 + 1;

  for (int kb = 0; kb < nkb; ++kb) {
    const int k0 = kb * 32;
    for (int ch = lane; ch < 32 * 16; ch += 32) {  // K tile via async
      int r = ch >> 4, cp = ch & 15;
      cp16_g2l(&Ks[r][cp * 8],
               &kbf[((size_t)(k0 + r) * kKVH + kvh) * kHD + cp * 8]);
    }
    for (int i = lane; i < 32 * 128; i += 32) {    // V transpose (VALU path)
      int r = i >> 7, cc = i & 127;
      Vst[cc][r] = vbf[((size_t)(k0 + r) * kKVH + kvh) * kHD + cc];
    }
    wait_async_lds();
    __syncthreads();

    // scores: two 16x16 tiles covering 32 keys
    v8f s0 = zero8(), s1 = zero8();
#pragma unroll
    for (int kd = 0; kd < 128; kd += 32) {
      v16bf a  = ld_frag(&Qs[0][kd], 136);
      v16bf b0 = ld_frag(&Ks[0][kd], 136);
      v16bf b1 = ld_frag(&Ks[16][kd], 136);
      s0 = wmma_bf16(a, b0, s0);
      s1 = wmma_bf16(a, b1, s1);
    }

    const int col0 = k0 + r0, col1 = col0 + 16;
    float fr[8];
#pragma unroll
    for (int r = 0; r < 8; ++r) {
      const int row_pos = qt * 16 + r + 8 * hf;
      float v0 = s0[r] * scale; if (col0 > row_pos) v0 = -1e30f;
      float v1 = s1[r] * scale; if (col1 > row_pos) v1 = -1e30f;
      float mx = fmaxf(v0, v1);
#pragma unroll
      for (int off = 8; off >= 1; off >>= 1) mx = fmaxf(mx, __shfl_xor(mx, off, 16));
      const float nm = fmaxf(mrow[r], mx);
      const float f  = __expf(mrow[r] - nm);
      const float p0 = __expf(v0 - nm);
      const float p1 = __expf(v1 - nm);
      float ps = p0 + p1;
#pragma unroll
      for (int off = 8; off >= 1; off >>= 1) ps += __shfl_xor(ps, off, 16);
      lsum[r] = lsum[r] * f + ps;
      mrow[r] = nm;
      fr[r]   = f;
      Ps[r + 8 * hf][r0]      = (__bf16)p0;
      Ps[r + 8 * hf][r0 + 16] = (__bf16)p1;
    }
#pragma unroll
    for (int nt = 0; nt < 8; ++nt)
#pragma unroll
      for (int r = 0; r < 8; ++r) o[nt][r] *= fr[r];
    __syncthreads();

    // P(16x32) @ V(32x128)
    v16bf a = ld_frag(&Ps[0][0], 36);
#pragma unroll
    for (int nt = 0; nt < 8; ++nt) {
      v16bf b = ld_frag(&Vst[nt * 16][0], 36);
      o[nt] = wmma_bf16(a, b, o[nt]);
    }
    __syncthreads();
  }

#pragma unroll
  for (int nt = 0; nt < 8; ++nt)
#pragma unroll
    for (int r = 0; r < 8; ++r) {
      const int row = qt * 16 + r + 8 * hf;
      const int col = nt * 16 + r0;
      attnbf[(size_t)row * kQD + hh * kHD + col] = (__bf16)(o[nt][r] / lsum[r]);
    }
}

// ---------------------------------------------------------------------------
// Router / MoE bookkeeping
// ---------------------------------------------------------------------------
__global__ __launch_bounds__(256) void router_topk_kernel(const float* __restrict__ rs,
                                                          const float* __restrict__ bias,
                                                          int* __restrict__ idx,
                                                          float* __restrict__ wts,
                                                          int* __restrict__ counts) {
  const int t = blockIdx.x * 256 + threadIdx.x;
  if (t >= kT) return;
  float s[kE], sb[kE];
#pragma unroll
  for (int e = 0; e < kE; ++e) {
    float v = 1.0f / (1.0f + __expf(-rs[(size_t)t * kE + e]));
    s[e]  = v;
    sb[e] = v + bias[e];
  }
  int isel[kTOPK]; float wsel[kTOPK]; float wsum = 0.0f;
#pragma unroll
  for (int k = 0; k < kTOPK; ++k) {
    int best = 0; float bv = -1e30f;
#pragma unroll
    for (int e = 0; e < kE; ++e)
      if (sb[e] > bv) { bv = sb[e]; best = e; }
    isel[k] = best; wsel[k] = s[best]; sb[best] = -1e30f; wsum += s[best];
  }
  const float inv = 1.0f / wsum;
#pragma unroll
  for (int k = 0; k < kTOPK; ++k) {
    idx[t * kTOPK + k] = isel[k];
    wts[t * kTOPK + k] = wsel[k] * inv;
    atomicAdd(&counts[isel[k]], 1);
  }
}

__global__ void scan_kernel(const int* __restrict__ counts, int* __restrict__ offs,
                            int* __restrict__ cursor) {
  if (threadIdx.x == 0) {
    int acc = 0;
    for (int e = 0; e < kE; ++e) { offs[e] = acc; acc += counts[e]; cursor[e] = 0; }
  }
}

__global__ __launch_bounds__(256) void fill_kernel(const int* __restrict__ idx,
                                                   const float* __restrict__ wts,
                                                   const int* __restrict__ offs,
                                                   int* __restrict__ cursor,
                                                   int* __restrict__ assign_tok,
                                                   float* __restrict__ assign_w) {
  const int g = blockIdx.x * 256 + threadIdx.x;
  if (g >= kA) return;
  const int e = idx[g];
  const int p = atomicAdd(&cursor[e], 1);
  const int a = offs[e] + p;
  assign_tok[a] = g >> 3;
  assign_w[a]   = wts[g];
}

// ---------------------------------------------------------------------------
// Grouped gate/up GEMM: per expert, A = gathered bf16 token rows [cnt x H],
// B = Wg[e]/Wu[e] (H x I). Grid (E, I/64). Block tile M=32, N=64; 4 waves:
// wave (mh, nh) owns 16x32 of BOTH g and u -> 4 WMMA / 5 fragment loads.
// ---------------------------------------------------------------------------
__global__ __launch_bounds__(128) void moe_gateup_kernel(const __bf16* __restrict__ h2,
                                                         const float* __restrict__ Wg,
                                                         const float* __restrict__ Wu,
                                                         const int* __restrict__ assign_tok,
                                                         const int* __restrict__ offs,
                                                         const int* __restrict__ counts,
                                                         float* __restrict__ gout,
                                                         float* __restrict__ uout) {
  __shared__ __bf16 As[32][40];
  __shared__ __bf16 Btg[64][40];
  __shared__ __bf16 Btu[64][40];
  __shared__ int toks[32];
  const int e = blockIdx.x;
  const int cnt = counts[e];
  if (cnt == 0) return;
  const int base = offs[e];
  const int tile_n = blockIdx.y * 64;
  const float* Bg = Wg + (size_t)e * kH * kI;
  const float* Bu = Wu + (size_t)e * kH * kI;
  const int tid = threadIdx.x, wave = tid >> 5, lane = tid & 31;
  const int wm = (wave >> 1) * 16, wn = (wave & 1) * 32;
  for (int m0 = 0; m0 < cnt; m0 += 32) {
    if (tid < 32) {
      int mr = m0 + tid;
      toks[tid] = assign_tok[base + (mr < cnt ? mr : cnt - 1)];  // clamp; padding discarded
    }
    __syncthreads();
    v8f cg[2] = {zero8(), zero8()}, cu[2] = {zero8(), zero8()};
    for (int k0 = 0; k0 < kH; k0 += 32) {
#pragma unroll
      for (int ch = tid; ch < 128; ch += 128) {  // A: 32 rows x 4 b128 chunks
        int r = ch >> 2, cp = ch & 3;
        cp16_g2l(&As[r][cp * 8], &h2[(size_t)toks[r] * kH + k0 + cp * 8]);
      }
      for (int i = tid; i < 2048; i += 128) {    // weights, coalesced over n
        int n = i & 63, kk = i >> 6;
        int gn = tile_n + n;
        Btg[n][kk] = (__bf16)Bg[(size_t)(k0 + kk) * kI + gn];
        Btu[n][kk] = (__bf16)Bu[(size_t)(k0 + kk) * kI + gn];
      }
      wait_async_lds();
      __syncthreads();
      v16bf a = ld_frag(&As[wm][0], 40);
#pragma unroll
      for (int j = 0; j < 2; ++j) {
        v16bf bg = ld_frag(&Btg[wn + j * 16][0], 40);
        v16bf bu = ld_frag(&Btu[wn + j * 16][0], 40);
        cg[j] = wmma_bf16(a, bg, cg[j]);
        cu[j] = wmma_bf16(a, bu, cu[j]);
      }
      __syncthreads();
    }
    const int r0 = lane & 15, hf = lane >> 4;
#pragma unroll
    for (int j = 0; j < 2; ++j)
#pragma unroll
      for (int r = 0; r < 8; ++r) {
        int mrow = m0 + wm + r + 8 * hf;
        if (mrow < cnt) {
          size_t row = (size_t)(base + mrow);
          int col = tile_n + wn + j * 16 + r0;
          gout[row * kI + col] = cg[j][r];
          uout[row * kI + col] = cu[j][r];
        }
      }
    __syncthreads();
  }
}

// ---------------------------------------------------------------------------
// Grouped down GEMM with weighted scatter: A = act bf16 [cnt x I] (assignment-
// ordered), B = Wd[e] (I x H); atomicAdd(w * val) into per-token accumulator.
// Block tile M=32, N=64; wave = 16x32 (2 WMMA / 3 fragment loads).
// ---------------------------------------------------------------------------
__global__ __launch_bounds__(128) void moe_down_kernel(const __bf16* __restrict__ act,
                                                       const float* __restrict__ Wd,
                                                       const int* __restrict__ assign_tok,
                                                       const float* __restrict__ assign_w,
                                                       const int* __restrict__ offs,
                                                       const int* __restrict__ counts,
                                                       float* __restrict__ moe_acc) {
  __shared__ __bf16 As[32][40];
  __shared__ __bf16 Bt[64][40];
  const int e = blockIdx.x;
  const int cnt = counts[e];
  if (cnt == 0) return;
  const int base = offs[e];
  const int tile_n = blockIdx.y * 64;
  const float* Bw = Wd + (size_t)e * kI * kH;
  const int tid = threadIdx.x, wave = tid >> 5, lane = tid & 31;
  const int wm = (wave >> 1) * 16, wn = (wave & 1) * 32;
  for (int m0 = 0; m0 < cnt; m0 += 32) {
    v8f c[2] = {zero8(), zero8()};
    for (int k0 = 0; k0 < kI; k0 += 32) {
#pragma unroll
      for (int ch = tid; ch < 128; ch += 128) {  // A: 32 rows x 4 b128 chunks
        int r = ch >> 2, cp = ch & 3;
        int mr = m0 + r; if (mr >= cnt) mr = cnt - 1;  // clamp; padding discarded
        cp16_g2l(&As[r][cp * 8], &act[(size_t)(base + mr) * kI + k0 + cp * 8]);
      }
      for (int i = tid; i < 2048; i += 128) {
        int n = i & 63, kk = i >> 6;
        Bt[n][kk] = (__bf16)Bw[(size_t)(k0 + kk) * kH + tile_n + n];
      }
      wait_async_lds();
      __syncthreads();
      v16bf a = ld_frag(&As[wm][0], 40);
#pragma unroll
      for (int j = 0; j < 2; ++j) {
        v16bf b = ld_frag(&Bt[wn + j * 16][0], 40);
        c[j] = wmma_bf16(a, b, c[j]);
      }
      __syncthreads();
    }
    const int r0 = lane & 15, hf = lane >> 4;
#pragma unroll
    for (int j = 0; j < 2; ++j)
#pragma unroll
      for (int r = 0; r < 8; ++r) {
        int mrow = m0 + wm + r + 8 * hf;
        if (mrow < cnt) {
          int a = base + mrow;
          int tk = assign_tok[a];
          float w = assign_w[a];
          atomicAdd(&moe_acc[(size_t)tk * kH + tile_n + wn + j * 16 + r0], w * c[j][r]);
        }
      }
  }
}

// ---------------------------------------------------------------------------
// Host orchestration
// ---------------------------------------------------------------------------
extern "C" void kernel_launch(void* const* d_in, const int* in_sizes, int n_in,
                              void* d_out, int out_size, void* d_ws, size_t ws_size,
                              hipStream_t stream) {
  const float* x     = (const float*)d_in[0];
  const float* ln1   = (const float*)d_in[1];
  const float* ln2   = (const float*)d_in[2];
  const float* Wq    = (const float*)d_in[3];
  const float* Wk    = (const float*)d_in[4];
  const float* Wv    = (const float*)d_in[5];
  const float* Wo    = (const float*)d_in[6];
  const float* Wr    = (const float*)d_in[7];
  const float* ebias = (const float*)d_in[8];
  const float* Wg    = (const float*)d_in[9];
  const float* Wu    = (const float*)d_in[10];
  const float* Wd    = (const float*)d_in[11];
  const float* Wsg   = (const float*)d_in[12];
  const float* Wsu   = (const float*)d_in[13];
  const float* Wsd   = (const float*)d_in[14];
  float* out = (float*)d_out;

  char* ws = (char*)d_ws;
  size_t off = 0;
  auto alloc = [&](size_t bytes) -> void* {
    off = (off + 255) & ~(size_t)255;
    void* p = ws + off;
    off += bytes;
    return p;
  };

  __bf16* h1bf   = (__bf16*)alloc((size_t)kT * kH * 2);
  float*  qf     = (float*) alloc((size_t)kT * kQD * 4);
  float*  kf     = (float*) alloc((size_t)kT * kKD * 4);
  float*  vf     = (float*) alloc((size_t)kT * kKD * 4);
  __bf16* qbf    = (__bf16*)alloc((size_t)kT * kQD * 2);
  __bf16* kbf    = (__bf16*)alloc((size_t)kT * kKD * 2);
  __bf16* vbf    = (__bf16*)alloc((size_t)kT * kKD * 2);
  __bf16* attnbf = (__bf16*)alloc((size_t)kT * kQD * 2);
  float*  oproj  = (float*) alloc((size_t)kT * kH * 4);
  float*  x1     = (float*) alloc((size_t)kT * kH * 4);
  __bf16* h2bf   = (__bf16*)alloc((size_t)kT * kH * 2);
  float*  rscore = (float*) alloc((size_t)kT * kE * 4);
  int*    idxb   = (int*)   alloc((size_t)kA * 4);
  float*  wtsb   = (float*) alloc((size_t)kA * 4);
  int*    counts = (int*)   alloc(kE * 4);
  int*    offsb  = (int*)   alloc(kE * 4);
  int*    cursor = (int*)   alloc(kE * 4);
  int*    atok   = (int*)   alloc((size_t)kA * 4);
  float*  awt    = (float*) alloc((size_t)kA * 4);
  float*  gbuf   = (float*) alloc((size_t)kA * kI * 4);
  float*  ubuf   = (float*) alloc((size_t)kA * kI * 4);
  __bf16* actbf  = (__bf16*)alloc((size_t)kA * kI * 2);
  float*  moeacc = (float*) alloc((size_t)kT * kH * 4);
  float*  gs     = (float*) alloc((size_t)kT * kI * 4);
  float*  us     = (float*) alloc((size_t)kT * kI * 4);
  __bf16* actsbf = (__bf16*)alloc((size_t)kT * kI * 2);

  // zero accumulators / counters
  zero_kernel<<<(kT * kH + 255) / 256, 256, 0, stream>>>((unsigned*)moeacc, kT * kH);
  zero_kernel<<<1, 256, 0, stream>>>((unsigned*)counts, kE);

  // ---- attention ----
  rmsnorm_kernel<<<kT, 256, 0, stream>>>(x, ln1, h1bf);
  gemm_bf16_kernel<false><<<dim3(kQD / 64, kT / 64), 128, 0, stream>>>(h1bf, Wq, qf, kT, kQD, kH);
  gemm_bf16_kernel<false><<<dim3(kKD / 64, kT / 64), 128, 0, stream>>>(h1bf, Wk, kf, kT, kKD, kH);
  gemm_bf16_kernel<false><<<dim3(kKD / 64, kT / 64), 128, 0, stream>>>(h1bf, Wv, vf, kT, kKD, kH);
  rope_kernel<<<dim3(kT, kNH), 64, 0, stream>>>(qf, kf, vf, qbf, kbf, vbf);
  attn_kernel<<<dim3(kT / 16, kNH), 32, 0, stream>>>(qbf, kbf, vbf, attnbf);
  gemm_bf16_kernel<false><<<dim3(kH / 64, kT / 64), 128, 0, stream>>>(attnbf, Wo, oproj, kT, kH, kQD);
  add_kernel<<<(kT * kH + 255) / 256, 256, 0, stream>>>(x, oproj, x1, kT * kH);

  // ---- MoE FFN ----
  rmsnorm_kernel<<<kT, 256, 0, stream>>>(x1, ln2, h2bf);
  gemm_bf16_kernel<false><<<dim3(1, kT / 64), 128, 0, stream>>>(h2bf, Wr, rscore, kT, kE, kH);
  router_topk_kernel<<<(kT + 255) / 256, 256, 0, stream>>>(rscore, ebias, idxb, wtsb, counts);
  scan_kernel<<<1, 32, 0, stream>>>(counts, offsb, cursor);
  fill_kernel<<<(kA + 255) / 256, 256, 0, stream>>>(idxb, wtsb, offsb, cursor, atok, awt);
  moe_gateup_kernel<<<dim3(kE, kI / 64), 128, 0, stream>>>(h2bf, Wg, Wu, atok, offsb, counts, gbuf, ubuf);
  silu_mul_kernel<<<((int)((size_t)kA * kI) + 255) / 256, 256, 0, stream>>>(gbuf, ubuf, actbf, kA * kI);
  moe_down_kernel<<<dim3(kE, kH / 64), 128, 0, stream>>>(actbf, Wd, atok, awt, offsb, counts, moeacc);

  // shared expert
  gemm_bf16_kernel<false><<<dim3(kI / 64, kT / 64), 128, 0, stream>>>(h2bf, Wsg, gs, kT, kI, kH);
  gemm_bf16_kernel<false><<<dim3(kI / 64, kT / 64), 128, 0, stream>>>(h2bf, Wsu, us, kT, kI, kH);
  silu_mul_kernel<<<(kT * kI + 255) / 256, 256, 0, stream>>>(gs, us, actsbf, kT * kI);
  gemm_bf16_kernel<true><<<dim3(kH / 64, kT / 64), 128, 0, stream>>>(actsbf, Wsd, moeacc, kT, kH, kI);

  // final residual
  add_kernel<<<(kT * kH + 255) / 256, 256, 0, stream>>>(x1, moeacc, out, kT * kH);
}
